// NEMDO1_23416161697922
// MI455X (gfx1250) — compile-verified, matrix operand order checked by reading, not run on hardware
//
#include <hip/hip_runtime.h>
#include <hip/hip_bf16.h>
#include <cstdint>

#define N_NODES 50000
#define N_EDGES 800000
#define EDIM 128
#define NLAYERS 3
#define WSTRIDE 136  // bf16 row stride in LDS -> 272B rows, conflict-free b128 reads

typedef __bf16 bf16_t;
typedef __attribute__((ext_vector_type(16))) __bf16 v16bf;
typedef __attribute__((ext_vector_type(8)))  float  v8f;
typedef int v4i_vs __attribute__((__vector_size__(16)));

union ABfrag { v16bf v; uint4 u[2]; };
union B4 { uint2 u; bf16_t b[4]; };

// ---- branch-free tanh: v_exp_f32 + v_rcp_f32, no EXEC divergence ----------
__device__ __forceinline__ float fast_tanh(float x) {
  float e = __expf(-2.0f * fabsf(x));
  float y = (1.0f - e) * __builtin_amdgcn_rcpf(1.0f + e);
  return copysignf(y, x);
}

// monotonic float<->uint transform so atomicMax(u32) orders like float
__device__ __forceinline__ unsigned ftrans(float f) {
  unsigned u = __float_as_uint(f);
  return (u & 0x80000000u) ? ~u : (u | 0x80000000u);
}
__device__ __forceinline__ float finv(unsigned u) {
  u = (u & 0x80000000u) ? (u & 0x7fffffffu) : ~u;
  return __uint_as_float(u);
}

// ---- stage one 128x128 bf16 matrix into LDS (stride WSTRIDE) ---------------
// Uses CDNA5 async global->LDS DMA when the builtin is available.
#if defined(__gfx1250__) && __has_builtin(__builtin_amdgcn_global_load_async_to_lds_b128)
#define HAVE_ASYNC_LDS 1
#else
#define HAVE_ASYNC_LDS 0
#endif

__device__ __forceinline__ void stage_weights(const bf16_t* __restrict__ W,
                                              bf16_t* lds_w, int tid) {
  // 128 rows x 16 chunks of 16B; 128 threads -> 16 chunks each
  for (int i = tid; i < 128 * 16; i += 128) {
    const int r = i >> 4;
    const int c = (i & 15) * 8;
#if HAVE_ASYNC_LDS
    bf16_t* wp = const_cast<bf16_t*>(W) + (size_t)r * EDIM + c;
    __builtin_amdgcn_global_load_async_to_lds_b128(
        (__attribute__((address_space(1))) v4i_vs*)wp,
        (__attribute__((address_space(3))) v4i_vs*)(lds_w + r * WSTRIDE + c), 0, 0);
#else
    *(uint4*)(lds_w + r * WSTRIDE + c) = *(const uint4*)(W + (size_t)r * EDIM + c);
#endif
  }
#if HAVE_ASYNC_LDS
#if __has_builtin(__builtin_amdgcn_s_wait_asynccnt)
  __builtin_amdgcn_s_wait_asynccnt(0);
#else
  asm volatile("s_wait_asynccnt 0x0" ::: "memory");
#endif
#endif
  __syncthreads();
}

// ---------------- weight transpose: Wt[l][n][k] = W[l][k][n], f32 -> bf16 ----
__global__ void transpose_wt_kernel(const float* __restrict__ W,
                                    bf16_t* __restrict__ Wt,
                                    int K, int N, int total) {
  int idx = blockIdx.x * blockDim.x + threadIdx.x;
  if (idx >= total) return;
  int per = K * N;
  int l = idx / per;
  int r = idx - l * per;
  int n = r / K;
  int k = r - n * K;
  Wt[idx] = (bf16_t)W[(size_t)l * per + (size_t)k * N + n];
}

// ---------------- encoder: h = tanh(nf @ enc_W + b) ------------------------
__global__ void encode_kernel(const float* __restrict__ nf,
                              const float* __restrict__ encW,
                              const float* __restrict__ encb,
                              float* __restrict__ h, bf16_t* __restrict__ hb,
                              int total) {
  int idx = blockIdx.x * blockDim.x + threadIdx.x;
  if (idx >= total) return;
  int i = idx >> 7, c = idx & 127;
  float val = fast_tanh(nf[i * 2] * encW[c] + nf[i * 2 + 1] * encW[EDIM + c] + encb[c]);
  h[idx] = val;
  hb[idx] = (bf16_t)val;
}

// ---------------- fused edge kernel: mj -> gate, v (WMMA bf16) --------------
// 4 waves/block, one 16-edge tile per wave; weights phase-staged in LDS.
__global__ __launch_bounds__(128)
void edge_fused_kernel(const bf16_t* __restrict__ hb,
                       const int* __restrict__ src,
                       const bf16_t* __restrict__ msgWt,
                       const bf16_t* __restrict__ gateWt,
                       const bf16_t* __restrict__ attWt,
                       const float* __restrict__ msgb,
                       const float* __restrict__ gateb,
                       const float* __restrict__ attb,
                       float* __restrict__ gate_out,
                       bf16_t* __restrict__ v_out) {
  __shared__ __align__(16) bf16_t lds_w[128 * WSTRIDE];       // 34816 B
  __shared__ __align__(16) bf16_t lds_mj[4][16 * WSTRIDE];    // 17408 B
  const int tid  = threadIdx.x;
  const int wave = tid >> 5;
  const int lane = tid & 31;
  const int ebase = (blockIdx.x * 4 + wave) * 16;  // grid exact: no tail
  const int row  = lane & 15;
  const int half = lane >> 4;

  // warm L2 for the later weight matrices while we work on msg
  __builtin_prefetch(gateWt + (size_t)tid * EDIM, 0, 1);
  __builtin_prefetch(attWt + (size_t)tid * EDIM, 0, 1);

  // ---- gather A fragments: rows = h[src[e]], K = 128 (4 wmma K-steps) ----
  const bf16_t* arow = hb + (size_t)src[ebase + row] * EDIM;
  ABfrag a[4];
#pragma unroll
  for (int ks = 0; ks < 4; ++ks) {
    const int k0 = ks * 32 + half * 8;
    a[ks].u[0] = *(const uint4*)(arow + k0);
    a[ks].u[1] = *(const uint4*)(arow + k0 + 16);
  }

  // ================= mj = tanh(h[src] @ msg_W + b) =================
  stage_weights(msgWt, lds_w, tid);
#pragma unroll
  for (int nt = 0; nt < 8; ++nt) {
    const int n = nt * 16 + row;
    const bf16_t* brow = &lds_w[n * WSTRIDE];
    v8f acc = {};
#pragma unroll
    for (int ks = 0; ks < 4; ++ks) {
      ABfrag b;
      const int k0 = ks * 32 + half * 8;
      b.u[0] = *(const uint4*)(brow + k0);
      b.u[1] = *(const uint4*)(brow + k0 + 16);
      acc = __builtin_amdgcn_wmma_f32_16x16x32_bf16(false, a[ks].v, false, b.v,
                                                    (short)0, acc, false, false);
    }
    const float bias = msgb[n];
#pragma unroll
    for (int r = 0; r < 8; ++r) {
      const int m = r + half * 8;
      lds_mj[wave][m * WSTRIDE + n] = (bf16_t)fast_tanh(acc[r] + bias);
    }
  }
  __syncthreads();  // everyone done reading msg weights

  // ================= gate = mj @ gate_W + b =================
  stage_weights(gateWt, lds_w, tid);
  ABfrag am[4];
#pragma unroll
  for (int ks = 0; ks < 4; ++ks) {
    const int k0 = ks * 32 + half * 8;
    am[ks].u[0] = *(const uint4*)&lds_mj[wave][row * WSTRIDE + k0];
    am[ks].u[1] = *(const uint4*)&lds_mj[wave][row * WSTRIDE + k0 + 16];
  }
#pragma unroll
  for (int nt = 0; nt < 8; ++nt) {
    const int n = nt * 16 + row;
    const bf16_t* brow = &lds_w[n * WSTRIDE];
    v8f acc = {};
#pragma unroll
    for (int ks = 0; ks < 4; ++ks) {
      ABfrag b;
      const int k0 = ks * 32 + half * 8;
      b.u[0] = *(const uint4*)(brow + k0);
      b.u[1] = *(const uint4*)(brow + k0 + 16);
      acc = __builtin_amdgcn_wmma_f32_16x16x32_bf16(false, am[ks].v, false, b.v,
                                                    (short)0, acc, false, false);
    }
    const float bias = gateb[n];
#pragma unroll
    for (int r = 0; r < 8; ++r) {
      const int m = r + half * 8;
      gate_out[(size_t)(ebase + m) * EDIM + n] = acc[r] + bias;
    }
  }
  __syncthreads();  // everyone done reading gate weights

  // ================= v = tanh(mj @ att_W + b) =================
  stage_weights(attWt, lds_w, tid);
#pragma unroll
  for (int nt = 0; nt < 8; ++nt) {
    const int n = nt * 16 + row;
    const bf16_t* brow = &lds_w[n * WSTRIDE];
    v8f acc = {};
#pragma unroll
    for (int ks = 0; ks < 4; ++ks) {
      ABfrag b;
      const int k0 = ks * 32 + half * 8;
      b.u[0] = *(const uint4*)(brow + k0);
      b.u[1] = *(const uint4*)(brow + k0 + 16);
      acc = __builtin_amdgcn_wmma_f32_16x16x32_bf16(false, am[ks].v, false, b.v,
                                                    (short)0, acc, false, false);
    }
    const float bias = attb[n];
#pragma unroll
    for (int r = 0; r < 8; ++r) {
      const int m = r + half * 8;
      v_out[(size_t)(ebase + m) * EDIM + n] = (bf16_t)fast_tanh(acc[r] + bias);
    }
  }
}

// ---------------- segment softmax + aggregation (4 channels/thread) ---------
__global__ void seg_init_kernel(unsigned* __restrict__ m_u, float* __restrict__ s,
                                float* __restrict__ aggr, int total) {
  int idx = blockIdx.x * blockDim.x + threadIdx.x;
  if (idx >= total) return;
  m_u[idx] = 0u;  // ftrans(-inf) lower bound
  s[idx] = 0.f;
  aggr[idx] = 0.f;
}

__global__ void segmax_kernel(const float* __restrict__ gate,
                              const int* __restrict__ dst,
                              unsigned* __restrict__ m_u) {
  int idx = blockIdx.x * blockDim.x + threadIdx.x;  // N_EDGES*32 threads
  int e = idx >> 5, q = (idx & 31) * 4;
  float4 g = *(const float4*)(gate + (size_t)e * EDIM + q);
  unsigned* mp = m_u + (size_t)dst[e] * EDIM + q;
  atomicMax(mp + 0, ftrans(g.x));
  atomicMax(mp + 1, ftrans(g.y));
  atomicMax(mp + 2, ftrans(g.z));
  atomicMax(mp + 3, ftrans(g.w));
}

__global__ void expsum_kernel(float* __restrict__ gate,
                              const int* __restrict__ dst,
                              const unsigned* __restrict__ m_u,
                              float* __restrict__ s) {
  int idx = blockIdx.x * blockDim.x + threadIdx.x;
  int e = idx >> 5, q = (idx & 31) * 4;
  size_t gi = (size_t)e * EDIM + q;
  size_t di = (size_t)dst[e] * EDIM + q;
  float4 g = *(const float4*)(gate + gi);
  uint4 mu = *(const uint4*)(m_u + di);
  float4 ev;
  ev.x = __expf(g.x - finv(mu.x));
  ev.y = __expf(g.y - finv(mu.y));
  ev.z = __expf(g.z - finv(mu.z));
  ev.w = __expf(g.w - finv(mu.w));
  *(float4*)(gate + gi) = ev;
  atomicAdd(s + di + 0, ev.x);
  atomicAdd(s + di + 1, ev.y);
  atomicAdd(s + di + 2, ev.z);
  atomicAdd(s + di + 3, ev.w);
}

__global__ void aggr_kernel(const float* __restrict__ gate,
                            const int* __restrict__ dst,
                            const float* __restrict__ s,
                            const bf16_t* __restrict__ vbuf,
                            float* __restrict__ aggr) {
  int idx = blockIdx.x * blockDim.x + threadIdx.x;
  int e = idx >> 5, q = (idx & 31) * 4;
  size_t gi = (size_t)e * EDIM + q;
  size_t di = (size_t)dst[e] * EDIM + q;
  float4 g = *(const float4*)(gate + gi);
  float4 sv = *(const float4*)(s + di);
  B4 vb;
  vb.u = *(const uint2*)(vbuf + gi);
  atomicAdd(aggr + di + 0, g.x * __builtin_amdgcn_rcpf(sv.x + 1e-16f) * (float)vb.b[0]);
  atomicAdd(aggr + di + 1, g.y * __builtin_amdgcn_rcpf(sv.y + 1e-16f) * (float)vb.b[1]);
  atomicAdd(aggr + di + 2, g.z * __builtin_amdgcn_rcpf(sv.z + 1e-16f) * (float)vb.b[2]);
  atomicAdd(aggr + di + 3, g.w * __builtin_amdgcn_rcpf(sv.w + 1e-16f) * (float)vb.b[3]);
}

__global__ void tobf_kernel(const float* __restrict__ x, bf16_t* __restrict__ y, int total) {
  int idx = blockIdx.x * blockDim.x + threadIdx.x;
  if (idx < total) y[idx] = (bf16_t)x[idx];
}

// ---------------- node update: h = tanh([h, aggr] @ upd_W + b) (WMMA) -------
__global__ __launch_bounds__(256)
void node_update_kernel(bf16_t* __restrict__ hb,
                        const bf16_t* __restrict__ ab,
                        const bf16_t* __restrict__ updWt,
                        const float* __restrict__ updb,
                        float* __restrict__ h) {
  const int wave = threadIdx.x >> 5;
  const int lane = threadIdx.x & 31;
  const int tile = blockIdx.x * 8 + wave;
  const int nbase = tile * 16;
  if (nbase >= N_NODES) return;
  const int row  = lane & 15;
  const int half = lane >> 4;

  const bf16_t* h_row = hb + (size_t)(nbase + row) * EDIM;
  const bf16_t* a_row = ab + (size_t)(nbase + row) * EDIM;

  ABfrag a[8];  // K = 256: ks 0..3 from h, 4..7 from aggr
#pragma unroll
  for (int ks = 0; ks < 4; ++ks) {
    const int k0 = ks * 32 + half * 8;
    a[ks].u[0]     = *(const uint4*)(h_row + k0);
    a[ks].u[1]     = *(const uint4*)(h_row + k0 + 16);
    a[ks + 4].u[0] = *(const uint4*)(a_row + k0);
    a[ks + 4].u[1] = *(const uint4*)(a_row + k0 + 16);
  }

#pragma unroll
  for (int nt = 0; nt < 8; ++nt) {
    const int n = nt * 16 + row;
    const bf16_t* brow = updWt + (size_t)n * (2 * EDIM);
    v8f acc = {};
#pragma unroll
    for (int ks = 0; ks < 8; ++ks) {
      ABfrag b;
      const int k0 = ks * 32 + half * 8;
      b.u[0] = *(const uint4*)(brow + k0);
      b.u[1] = *(const uint4*)(brow + k0 + 16);
      acc = __builtin_amdgcn_wmma_f32_16x16x32_bf16(false, a[ks].v, false, b.v,
                                                    (short)0, acc, false, false);
    }
    const float bias = updb[n];
#pragma unroll
    for (int r = 0; r < 8; ++r) {
      const int m = r + half * 8;
      const float val = fast_tanh(acc[r] + bias);
      h[(size_t)(nbase + m) * EDIM + n] = val;
      hb[(size_t)(nbase + m) * EDIM + n] = (bf16_t)val;
    }
  }
}

// ---------------- decoder: out = tanh(h @ d1) @ d2 --------------------------
__global__ __launch_bounds__(64)
void decoder_kernel(const float* __restrict__ h,
                    const float* __restrict__ d1W, const float* __restrict__ d1b,
                    const float* __restrict__ d2W, const float* __restrict__ d2b,
                    float* __restrict__ out) {
  __shared__ float hrow[EDIM];
  __shared__ float mid[64];
  const int node = blockIdx.x;
  const int t = threadIdx.x;
  hrow[t]      = h[(size_t)node * EDIM + t];
  hrow[t + 64] = h[(size_t)node * EDIM + 64 + t];
  __syncthreads();
  float acc = d1b[t];
#pragma unroll 4
  for (int k = 0; k < EDIM; ++k) acc += hrow[k] * d1W[k * 64 + t];
  mid[t] = fast_tanh(acc);
  __syncthreads();
  if (t == 0) {
    float o = d2b[0];
#pragma unroll 4
    for (int k = 0; k < 64; ++k) o += mid[k] * d2W[k];
    out[node] = o;
  }
}

extern "C" void kernel_launch(void* const* d_in, const int* in_sizes, int n_in,
                              void* d_out, int out_size, void* d_ws, size_t ws_size,
                              hipStream_t stream) {
  (void)in_sizes; (void)n_in; (void)out_size; (void)ws_size;
  const float* node_feature = (const float*)d_in[0];
  const int*   edge_index   = (const int*)d_in[1];
  // d_in[2] = batch (unused: single graph)
  const float* enc_W  = (const float*)d_in[3];
  const float* enc_b  = (const float*)d_in[4];
  const float* gate_W = (const float*)d_in[5];
  const float* gate_b = (const float*)d_in[6];
  const float* att_W  = (const float*)d_in[7];
  const float* att_b  = (const float*)d_in[8];
  const float* msg_W  = (const float*)d_in[9];
  const float* msg_b  = (const float*)d_in[10];
  const float* upd_W  = (const float*)d_in[11];
  const float* upd_b  = (const float*)d_in[12];
  const float* dec1_W = (const float*)d_in[13];
  const float* dec1_b = (const float*)d_in[14];
  const float* dec2_W = (const float*)d_in[15];
  const float* dec2_b = (const float*)d_in[16];
  const int* src = edge_index;
  const int* dst = edge_index + N_EDGES;

  char* ws = (char*)d_ws;
  size_t off = 0;
  auto alloc = [&](size_t bytes) -> void* {
    void* p = ws + off;
    off = (off + bytes + 255) & ~(size_t)255;
    return p;
  };
  float*    h       = (float*)alloc((size_t)N_NODES * EDIM * 4);
  bf16_t*   hb      = (bf16_t*)alloc((size_t)N_NODES * EDIM * 2);
  float*    aggr    = (float*)alloc((size_t)N_NODES * EDIM * 4);
  bf16_t*   aggr_bf = (bf16_t*)alloc((size_t)N_NODES * EDIM * 2);
  unsigned* m_u     = (unsigned*)alloc((size_t)N_NODES * EDIM * 4);
  float*    ssum    = (float*)alloc((size_t)N_NODES * EDIM * 4);
  float*    gate    = (float*)alloc((size_t)N_EDGES * EDIM * 4);
  bf16_t*   vbuf    = (bf16_t*)alloc((size_t)N_EDGES * EDIM * 2);
  bf16_t*   msgWt   = (bf16_t*)alloc((size_t)NLAYERS * EDIM * EDIM * 2);
  bf16_t*   gateWt  = (bf16_t*)alloc((size_t)NLAYERS * EDIM * EDIM * 2);
  bf16_t*   attWt   = (bf16_t*)alloc((size_t)NLAYERS * EDIM * EDIM * 2);
  bf16_t*   updWt   = (bf16_t*)alloc((size_t)NLAYERS * 2 * EDIM * EDIM * 2);

  // 1) transpose weights to bf16 [n][k]
  {
    int total = NLAYERS * EDIM * EDIM;
    transpose_wt_kernel<<<(total + 255) / 256, 256, 0, stream>>>(msg_W, msgWt, EDIM, EDIM, total);
    transpose_wt_kernel<<<(total + 255) / 256, 256, 0, stream>>>(gate_W, gateWt, EDIM, EDIM, total);
    transpose_wt_kernel<<<(total + 255) / 256, 256, 0, stream>>>(att_W, attWt, EDIM, EDIM, total);
    int totU = NLAYERS * 2 * EDIM * EDIM;
    transpose_wt_kernel<<<(totU + 255) / 256, 256, 0, stream>>>(upd_W, updWt, 2 * EDIM, EDIM, totU);
  }
  // 2) encoder
  {
    int total = N_NODES * EDIM;
    encode_kernel<<<(total + 255) / 256, 256, 0, stream>>>(node_feature, enc_W, enc_b, h, hb, total);
  }
  // 3) message-passing layers
  const int ntot = N_NODES * EDIM;                 // 6.4M
  const int eblocks4 = (N_EDGES * 32) / 256;       // 100000 (exact, 4 ch/thread)
  for (int l = 0; l < NLAYERS; ++l) {
    edge_fused_kernel<<<N_EDGES / (16 * 4), 128, 0, stream>>>(
        hb, src,
        msgWt  + (size_t)l * EDIM * EDIM,
        gateWt + (size_t)l * EDIM * EDIM,
        attWt  + (size_t)l * EDIM * EDIM,
        msg_b + l * EDIM, gate_b + l * EDIM, att_b + l * EDIM,
        gate, vbuf);
    seg_init_kernel<<<(ntot + 255) / 256, 256, 0, stream>>>(m_u, ssum, aggr, ntot);
    segmax_kernel<<<eblocks4, 256, 0, stream>>>(gate, dst, m_u);
    expsum_kernel<<<eblocks4, 256, 0, stream>>>(gate, dst, m_u, ssum);
    aggr_kernel<<<eblocks4, 256, 0, stream>>>(gate, dst, ssum, vbuf, aggr);
    tobf_kernel<<<(ntot + 255) / 256, 256, 0, stream>>>(aggr, aggr_bf, ntot);
    node_update_kernel<<<(N_NODES / 16 + 7) / 8, 256, 0, stream>>>(
        hb, aggr_bf, updWt + (size_t)l * 2 * EDIM * EDIM, upd_b + l * EDIM, h);
  }
  // 4) decoder
  decoder_kernel<<<N_NODES, 64, 0, stream>>>(h, dec1_W, dec1_b, dec2_W, dec2_b, (float*)d_out);
}
